// MemTransformerLM_57200374448503
// MI455X (gfx1250) — matrix operand order repaired and checked by reading, not compiled, not run on hardware
//
#include <hip/hip_runtime.h>
#include <cstdint>
#include <cstddef>

// ---------------- model constants ----------------
#define NLAYER 8
#define NHEAD  16
#define DMODEL 1024
#define DHEAD  64
#define DINNER 4096
#define QLEN   512
#define MLEN   512
#define KLEN   1024   /* QLEN + MLEN */
#define BSZ    16
#define HD     1024   /* NHEAD*DHEAD */
#define HD3    3072

typedef __attribute__((ext_vector_type(16))) __bf16 bf16x16;
typedef __attribute__((ext_vector_type(8)))  float  floatx8;

// ---------------- CDNA5 async global->LDS copy (ASYNCcnt) ----------------
#if defined(__AMDGCN__) && __has_builtin(__builtin_amdgcn_global_load_async_to_lds_b128)
#define HAVE_ASYNC_LDS 1
#else
#define HAVE_ASYNC_LDS 0
#endif

typedef int v4i __attribute__((vector_size(16)));
typedef __attribute__((address_space(1))) v4i* gptr_v4i;
typedef __attribute__((address_space(3))) v4i* lptr_v4i;

__device__ __forceinline__ void async_copy_b128(const __bf16* gsrc, __bf16* ldst)
{
#if HAVE_ASYNC_LDS
    __builtin_amdgcn_global_load_async_to_lds_b128(
        (gptr_v4i)(unsigned long long)(uintptr_t)gsrc,
        (lptr_v4i)(unsigned int)(uintptr_t)ldst,
        0, 0);
#else
    *(uint4*)ldst = *(const uint4*)gsrc;
#endif
}

__device__ __forceinline__ void wait_async0()
{
#if HAVE_ASYNC_LDS
#if __has_builtin(__builtin_amdgcn_s_wait_asynccnt)
    __builtin_amdgcn_s_wait_asynccnt(0);
#else
    asm volatile("s_wait_asynccnt 0" ::: "memory");
#endif
#endif
}

// =====================================================================
// WMMA GEMM:  out[M,N] = A[M,K](bf16,row-major) x W[N,K]^T (bf16,row-major)
// optional bias[N] + ReLU; writes fp32 (outf) or bf16 (outb).
// Block 256 thr = 8 waves; tile 128x128; K-step 32; each wave: 32x64
// (8 WMMAs / K-step).  Double-buffered LDS staging via async-to-LDS.
// Requires M%128==0, N%128==0, K%32==0.
// =====================================================================
#define LSTR 48  // padded LDS row stride (bf16 elems); 96B rows keep 16B chunks aligned

__global__ __launch_bounds__(256) void gemm_bf16_kernel(
    const __bf16* __restrict__ A, const __bf16* __restrict__ W,
    float* __restrict__ outf, __bf16* __restrict__ outb,
    int M, int N, int K,
    const float* __restrict__ bias, int relu)
{
    __shared__ __bf16 As[2][128 * LSTR];
    __shared__ __bf16 Bs[2][128 * LSTR];

    const int tid  = threadIdx.x;
    const int lane = tid & 31;
    const int wave = tid >> 5;
    const int L = lane & 15;         // lane within half-wave
    const int g = lane >> 4;         // half-wave select
    const int m0 = blockIdx.y * 128;
    const int n0 = blockIdx.x * 128;
    const int wm = (wave & 3) * 32;  // wave's 32-row slab
    const int wn = (wave >> 2) * 64; // wave's 64-col slab (4 x 16-col tiles)

    floatx8 acc[2][4];
#pragma unroll
    for (int i = 0; i < 2; ++i)
#pragma unroll
        for (int t = 0; t < 4; ++t) acc[i][t] = (floatx8){};

    auto stage = [&](int kt, int buf) {
        const int k0 = kt * 32;
#pragma unroll
        for (int it = 0; it < 2; ++it) {
            int idx = tid + it * 256;
            int row = idx >> 2, ch = idx & 3;    // 128 rows x 4 x 16B chunks
            async_copy_b128(A + (size_t)(m0 + row) * K + k0 + ch * 8,
                            &As[buf][row * LSTR + ch * 8]);
            async_copy_b128(W + (size_t)(n0 + row) * K + k0 + ch * 8,
                            &Bs[buf][row * LSTR + ch * 8]);
        }
    };

    const int nk = K / 32;
    stage(0, 0);
    for (int t = 0; t < nk; ++t) {
        const int buf = t & 1;
        wait_async0();
        __syncthreads();                 // tile t visible to all waves
        if (t + 1 < nk) stage(t + 1, buf ^ 1);

        bf16x16 a0, a1, b[4];
#pragma unroll
        for (int e = 0; e < 8; ++e) {
            a0[e]     = As[buf][(wm + L) * LSTR + 8 * g + e];
            a0[e + 8] = As[buf][(wm + L) * LSTR + 16 + 8 * g + e];
            a1[e]     = As[buf][(wm + 16 + L) * LSTR + 8 * g + e];
            a1[e + 8] = As[buf][(wm + 16 + L) * LSTR + 16 + 8 * g + e];
        }
#pragma unroll
        for (int c = 0; c < 4; ++c)
#pragma unroll
            for (int e = 0; e < 16; ++e)
                b[c][e] = Bs[buf][(wn + 16 * c + L) * LSTR + 8 * g + e + (e & 8)];

#pragma unroll
        for (int c = 0; c < 4; ++c) {
            acc[0][c] = __builtin_amdgcn_wmma_f32_16x16x32_bf16(false, a0, false, b[c], (short)0, acc[0][c], false, false);
            acc[1][c] = __builtin_amdgcn_wmma_f32_16x16x32_bf16(false, a1, false, b[c], (short)0, acc[1][c], false, false);
        }
        __syncthreads();                 // all waves done reading buf
    }

    // epilogue: D element (M = 8g + r, N = L) within each 16x16 tile
#pragma unroll
    for (int i = 0; i < 2; ++i) {
#pragma unroll
        for (int c = 0; c < 4; ++c) {
            int gn = n0 + wn + 16 * c + L;
            float bv = bias ? bias[gn] : 0.f;
#pragma unroll
            for (int r = 0; r < 8; ++r) {
                int gm = m0 + wm + 16 * i + 8 * g + r;
                float v = acc[i][c][r] + bv;
                if (relu) v = fmaxf(v, 0.f);
                if (outb) outb[(size_t)gm * N + gn] = (__bf16)v;
                else      outf[(size_t)gm * N + gn] = v;
            }
        }
    }
}

// =====================================================================
// Attention: one block per (q-tile of 16, head, batch).
// AC = (q + r_w_bias) k^T and BDraw = (q + r_r_bias) rk^T via WMMA,
// rel-shift applied analytically, masked softmax fp32, P.V via WMMA.
// heads/rk are bf16; output vec is bf16 (feeds o-proj GEMM directly).
// =====================================================================
__global__ __launch_bounds__(256) void attn_kernel(
    const __bf16* __restrict__ heads,  // [KLEN, BSZ, 3*HD] bf16
    const __bf16* __restrict__ rk,     // [KLEN, HD] bf16
    const float* __restrict__ rwb,     // [HD]
    const float* __restrict__ rrb,     // [HD]
    __bf16* __restrict__ vec)          // [QLEN, BSZ, HD] bf16
{
    extern __shared__ char smem[];
    float*  AC_s  = (float*)smem;                 // 16*1024 f32 (later prob)
    float*  BD_s  = AC_s + 16 * 1024;             // 16*1024 f32
    float*  pv_s  = BD_s + 16 * 1024;             // 2*16*64 f32
    float*  red_s = pv_s + 2 * 16 * 64;           // 16*17 f32
    __bf16* qw_s  = (__bf16*)(red_s + 16 * 17);   // 16*64 bf16
    __bf16* qr_s  = qw_s + 16 * 64;               // 16*64 bf16

    const int qt  = blockIdx.x;   // 0..31
    const int n   = blockIdx.y;   // head
    const int b   = blockIdx.z;   // batch
    const int i0  = qt * 16;
    const int tid = threadIdx.x;
    const int lane = tid & 31;
    const int wave = tid >> 5;
    const int L = lane & 15;
    const int g = lane >> 4;

    // ---- load q tile, add biases ----
    for (int idx = tid; idx < 16 * 64; idx += 256) {
        int i = idx >> 6, d = idx & 63;
        float qv = (float)heads[(size_t)(MLEN + i0 + i) * (BSZ * HD3) + (size_t)b * HD3 + n * 64 + d];
        qw_s[i * 64 + d] = (__bf16)(qv + rwb[n * 64 + d]);
        qr_s[i * 64 + d] = (__bf16)(qv + rrb[n * 64 + d]);
    }
    __syncthreads();

    // ---- AC and raw BD over 64 j-tiles (8 per wave) ----
    for (int jt = wave; jt < 64; jt += 8) {
        int j0 = jt * 16;
        floatx8 accA = {}, accB = {};
        for (int k0 = 0; k0 < 64; k0 += 32) {
            bf16x16 aw, ar, bk, br;
#pragma unroll
            for (int e = 0; e < 8; ++e) {
                aw[e]     = qw_s[L * 64 + k0 + 8 * g + e];
                aw[e + 8] = qw_s[L * 64 + k0 + 16 + 8 * g + e];
                ar[e]     = qr_s[L * 64 + k0 + 8 * g + e];
                ar[e + 8] = qr_s[L * 64 + k0 + 16 + 8 * g + e];
            }
#pragma unroll
            for (int e = 0; e < 16; ++e) {
                int d = k0 + 8 * g + e + (e & 8);
                bk[e] = heads[(size_t)(j0 + L) * (BSZ * HD3) + (size_t)b * HD3 + HD + n * 64 + d];
                br[e] = rk[(size_t)(j0 + L) * HD + n * 64 + d];
            }
            accA = __builtin_amdgcn_wmma_f32_16x16x32_bf16(false, aw, false, bk, (short)0, accA, false, false);
            accB = __builtin_amdgcn_wmma_f32_16x16x32_bf16(false, ar, false, br, (short)0, accB, false, false);
        }
#pragma unroll
        for (int r = 0; r < 8; ++r) {
            AC_s[(8 * g + r) * 1024 + j0 + L] = accA[r];
            BD_s[(8 * g + r) * 1024 + j0 + L] = accB[r];
        }
    }
    __syncthreads();

    // ---- masked softmax with rel-shift: score(i,j)=s*(AC[i][j]+BD[i][j+511-i]) ----
    {
        const int ir = tid >> 4;          // row 0..15
        const int c  = tid & 15;          // 16 threads per row
        const int ig = i0 + ir;           // global q index
        const int shift = (QLEN - 1) - ig;
        const int jmax = ig + MLEN;       // mask: j <= i + mlen
        const float scale = 0.125f;       // 1/sqrt(64)

        float mx = -3.0e38f;
        for (int j = c; j < KLEN; j += 16)
            if (j <= jmax)
                mx = fmaxf(mx, scale * (AC_s[ir * 1024 + j] + BD_s[ir * 1024 + j + shift]));
        red_s[ir * 17 + c] = mx;
        __syncthreads();
        if (c == 0) {
            float m = red_s[ir * 17];
            for (int t = 1; t < 16; ++t) m = fmaxf(m, red_s[ir * 17 + t]);
            red_s[ir * 17 + 16] = m;
        }
        __syncthreads();
        float m = red_s[ir * 17 + 16];
        float ssum = 0.f;
        for (int j = c; j < KLEN; j += 16) {
            float p = 0.f;
            if (j <= jmax)
                p = __expf(scale * (AC_s[ir * 1024 + j] + BD_s[ir * 1024 + j + shift]) - m);
            AC_s[ir * 1024 + j] = p;
            ssum += p;
        }
        red_s[ir * 17 + c] = ssum;
        __syncthreads();
        if (c == 0) {
            float s2 = 0.f;
            for (int t = 0; t < 16; ++t) s2 += red_s[ir * 17 + t];
            red_s[ir * 17 + 16] = s2;
        }
        __syncthreads();
        float inv = 1.0f / red_s[ir * 17 + 16];
        for (int j = c; j < KLEN; j += 16) AC_s[ir * 1024 + j] *= inv;
    }
    __syncthreads();

    // ---- P.V : 16x64 output; 4 col-tiles x 2 K-halves across 8 waves ----
    {
        const int ct = wave & 3;   // d-tile (16 cols)
        const int kh = wave >> 2;  // K half (512)
        floatx8 acc = {};
        for (int j0 = kh * 512; j0 < kh * 512 + 512; j0 += 32) {
            bf16x16 a, bv;
#pragma unroll
            for (int e = 0; e < 8; ++e) {
                a[e]     = (__bf16)AC_s[L * 1024 + j0 + 8 * g + e];
                a[e + 8] = (__bf16)AC_s[L * 1024 + j0 + 16 + 8 * g + e];
            }
#pragma unroll
            for (int e = 0; e < 16; ++e) {
                int j = j0 + 8 * g + e + (e & 8);
                bv[e] = heads[(size_t)j * (BSZ * HD3) + (size_t)b * HD3 + 2 * HD + n * 64 + ct * 16 + L];
            }
            acc = __builtin_amdgcn_wmma_f32_16x16x32_bf16(false, a, false, bv, (short)0, acc, false, false);
        }
#pragma unroll
        for (int r = 0; r < 8; ++r)
            pv_s[(kh * 16 + 8 * g + r) * 64 + ct * 16 + L] = acc[r];
    }
    __syncthreads();

    for (int idx = tid; idx < 16 * 64; idx += 256) {
        int i = idx >> 6, d = idx & 63;
        vec[(size_t)(i0 + i) * (BSZ * HD) + (size_t)b * HD + n * 64 + d] =
            (__bf16)(pv_s[i * 64 + d] + pv_s[(16 + i) * 64 + d]);
    }
}

// =====================================================================
// Elementwise helpers
// =====================================================================
__global__ void cvt_bf16_kernel(const float* __restrict__ src, __bf16* __restrict__ dst, size_t n)
{
    size_t i = (size_t)blockIdx.x * 256 + threadIdx.x;
    size_t st = (size_t)gridDim.x * 256;
    for (; i < n; i += st) dst[i] = (__bf16)src[i];
}

__global__ void embed_kernel(const int* __restrict__ inp, const float* __restrict__ emb,
                             float* __restrict__ core)
{
    size_t idx = (size_t)blockIdx.x * 256 + threadIdx.x;  // QLEN*BSZ*DMODEL
    if (idx >= (size_t)QLEN * BSZ * DMODEL) return;
    size_t row = idx >> 10;
    int d = idx & 1023;
    int tok = inp[row];
    core[idx] = emb[(size_t)tok * DMODEL + d] * 32.0f;  // sqrt(1024)
}

__global__ void posemb_kernel(float* __restrict__ rf)
{
    size_t idx = (size_t)blockIdx.x * 256 + threadIdx.x;  // KLEN*DMODEL
    if (idx >= (size_t)KLEN * DMODEL) return;
    int j = (int)(idx >> 10), d = (int)(idx & 1023);
    float pos = (float)(KLEN - 1 - j);
    int t = (d < 512) ? d : d - 512;
    float freq = __expf(-(float)(2 * t) * (1.0f / 1024.f) * 9.210340371976184f); // ln(1e4)
    float ang = pos * freq;
    rf[idx] = (d < 512) ? sinf(ang) : cosf(ang);
}

__global__ void concat_cvt_kernel(const float* __restrict__ memsl, const float* __restrict__ core,
                                  __bf16* __restrict__ catb)
{
    const size_t half = (size_t)MLEN * BSZ * DMODEL;
    size_t idx = (size_t)blockIdx.x * 256 + threadIdx.x;
    size_t st = (size_t)gridDim.x * 256;
    for (; idx < 2 * half; idx += st) {
        float v = (idx < half) ? memsl[idx] : core[idx - half];
        catb[idx] = (__bf16)v;
    }
}

// out_row = LayerNorm(xa_row + xb_row)*g + b ; optional bf16 copy (outb)
__global__ __launch_bounds__(256) void add_ln_kernel(
    const float* __restrict__ xa, const float* __restrict__ xb,
    const float* __restrict__ gg, const float* __restrict__ bb,
    float* __restrict__ outf, __bf16* __restrict__ outb)
{
    __shared__ float red[256];
    const int row = blockIdx.x;
    const int tid = threadIdx.x;
    float v[4];
    float s = 0.f;
#pragma unroll
    for (int t = 0; t < 4; ++t) {
        int i = tid + t * 256;
        v[t] = xa[(size_t)row * 1024 + i] + xb[(size_t)row * 1024 + i];
        s += v[t];
    }
    red[tid] = s; __syncthreads();
    for (int st = 128; st > 0; st >>= 1) { if (tid < st) red[tid] += red[tid + st]; __syncthreads(); }
    float mean = red[0] * (1.f / 1024.f);
    __syncthreads();
    s = 0.f;
#pragma unroll
    for (int t = 0; t < 4; ++t) { float d = v[t] - mean; s += d * d; }
    red[tid] = s; __syncthreads();
    for (int st = 128; st > 0; st >>= 1) { if (tid < st) red[tid] += red[tid + st]; __syncthreads(); }
    float inv = rsqrtf(red[0] * (1.f / 1024.f) + 1e-5f);
#pragma unroll
    for (int t = 0; t < 4; ++t) {
        int i = tid + t * 256;
        float o = (v[t] - mean) * inv * gg[i] + bb[i];
        outf[(size_t)row * 1024 + i] = o;
        if (outb) outb[(size_t)row * 1024 + i] = (__bf16)o;
    }
}

// =====================================================================
// Host orchestration
// =====================================================================
extern "C" void kernel_launch(void* const* d_in, const int* in_sizes, int n_in,
                              void* d_out, int out_size, void* d_ws, size_t ws_size,
                              hipStream_t stream)
{
    (void)in_sizes; (void)n_in; (void)out_size; (void)ws_size;

    const int*   inp      = (const int*)d_in[0];
    const float* mems     = (const float*)d_in[1];
    const float* emb      = (const float*)d_in[2];
    const float* r_w_bias = (const float*)d_in[3];
    const float* r_r_bias = (const float*)d_in[4];
    const float* qkv_w    = (const float*)d_in[5];
    const float* r_w      = (const float*)d_in[6];
    const float* o_w      = (const float*)d_in[7];
    const float* ln1_g    = (const float*)d_in[8];
    const float* ln1_b    = (const float*)d_in[9];
    const float* ff_w1    = (const float*)d_in[10];
    const float* ff_b1    = (const float*)d_in[11];
    const float* ff_w2    = (const float*)d_in[12];
    const float* ff_b2    = (const float*)d_in[13];
    const float* ln2_g    = (const float*)d_in[14];
    const float* ln2_b    = (const float*)d_in[15];
    float* core = (float*)d_out;   // persistent activations live in d_out

    // element counts
    const size_t WQKV_E  = (size_t)NLAYER * HD3 * DMODEL;
    const size_t WR_E    = (size_t)NLAYER * HD * DMODEL;
    const size_t WO_E    = (size_t)NLAYER * DMODEL * HD;
    const size_t WF1_E   = (size_t)NLAYER * DINNER * DMODEL;
    const size_t WF2_E   = (size_t)NLAYER * DMODEL * DINNER;
    const size_t R_E     = (size_t)KLEN * DMODEL;
    const size_t CAT_E   = (size_t)KLEN * BSZ * DMODEL;
    const size_t HEADS_E = (size_t)KLEN * BSZ * HD3;
    const size_t RK_E    = (size_t)KLEN * HD;
    const size_t ACT_E   = (size_t)QLEN * BSZ * DMODEL;
    const size_t FFA_E   = (size_t)QLEN * BSZ * DINNER;

    // workspace partition
    char* ws = (char*)d_ws;
    size_t off = 0;
    auto alloc = [&](size_t nbytes) -> void* {
        void* p = (void*)(ws + off);
        off = (off + nbytes + 255) & ~(size_t)255;
        return p;
    };
    __bf16* wqkvb = (__bf16*)alloc(WQKV_E * 2);
    __bf16* wrb   = (__bf16*)alloc(WR_E * 2);
    __bf16* wob   = (__bf16*)alloc(WO_E * 2);
    __bf16* wff1b = (__bf16*)alloc(WF1_E * 2);
    __bf16* wff2b = (__bf16*)alloc(WF2_E * 2);
    float*  rf    = (float*)alloc(R_E * 4);
    __bf16* rb    = (__bf16*)alloc(R_E * 2);
    __bf16* catb  = (__bf16*)alloc(CAT_E * 2);
    __bf16* headsb= (__bf16*)alloc(HEADS_E * 2);
    __bf16* rkb   = (__bf16*)alloc(RK_E * 2);
    __bf16* vecb  = (__bf16*)alloc(ACT_E * 2);
    float*  attn_o= (float*)alloc(ACT_E * 4);
    float*  hbuf  = (float*)alloc(ACT_E * 4);
    __bf16* hb    = (__bf16*)alloc(ACT_E * 2);
    __bf16* ffab  = (__bf16*)alloc(FFA_E * 2);
    float*  ffo   = (float*)alloc(ACT_E * 4);

    // ---- weight / posemb conversion (bf16) ----
    cvt_bf16_kernel<<<4096, 256, 0, stream>>>(qkv_w, wqkvb, WQKV_E);
    cvt_bf16_kernel<<<4096, 256, 0, stream>>>(r_w,   wrb,   WR_E);
    cvt_bf16_kernel<<<4096, 256, 0, stream>>>(o_w,   wob,   WO_E);
    cvt_bf16_kernel<<<4096, 256, 0, stream>>>(ff_w1, wff1b, WF1_E);
    cvt_bf16_kernel<<<4096, 256, 0, stream>>>(ff_w2, wff2b, WF2_E);
    posemb_kernel<<<(unsigned)((R_E + 255) / 256), 256, 0, stream>>>(rf);
    cvt_bf16_kernel<<<2048, 256, 0, stream>>>(rf, rb, R_E);

    // ---- embedding ----
    embed_kernel<<<(unsigned)((ACT_E + 255) / 256), 256, 0, stream>>>(inp, emb, core);

    const size_t attn_lds = 16 * 1024 * 4 /*AC*/ + 16 * 1024 * 4 /*BD*/
                          + 2 * 16 * 64 * 4 /*pv*/ + 16 * 17 * 4 /*red*/
                          + 16 * 64 * 2 * 2 /*qw+qr*/;

    for (int l = 0; l < NLAYER; ++l) {
        const float* mems_l = mems + (size_t)l * MLEN * BSZ * DMODEL;

        // cat = [mems_l ; core] -> bf16
        concat_cvt_kernel<<<8192, 256, 0, stream>>>(mems_l, core, catb);

        // heads(bf16) = cat @ qkv_w[l]^T : [16384, 3072]
        gemm_bf16_kernel<<<dim3(HD3 / 128, (KLEN * BSZ) / 128), 256, 0, stream>>>(
            catb, wqkvb + (size_t)l * HD3 * DMODEL, nullptr, headsb,
            KLEN * BSZ, HD3, DMODEL, nullptr, 0);

        // rk(bf16) = r @ r_w[l]^T : [1024, 1024]
        gemm_bf16_kernel<<<dim3(HD / 128, KLEN / 128), 256, 0, stream>>>(
            rb, wrb + (size_t)l * HD * DMODEL, nullptr, rkb,
            KLEN, HD, DMODEL, nullptr, 0);

        // attention (WMMA AC/BD + softmax + WMMA P.V), bf16 out
        attn_kernel<<<dim3(QLEN / 16, NHEAD, BSZ), 256, attn_lds, stream>>>(
            headsb, rkb, r_w_bias, r_r_bias, vecb);

        // attn_out(f32) = vec @ o_w[l]^T
        gemm_bf16_kernel<<<dim3(DMODEL / 128, (QLEN * BSZ) / 128), 256, 0, stream>>>(
            vecb, wob + (size_t)l * DMODEL * HD, attn_o, nullptr,
            QLEN * BSZ, DMODEL, HD, nullptr, 0);

        // h = LN(core + attn_out)  (fp32 + bf16)
        add_ln_kernel<<<QLEN * BSZ, 256, 0, stream>>>(
            core, attn_o, ln1_g + (size_t)l * DMODEL, ln1_b + (size_t)l * DMODEL, hbuf, hb);

        // ff1(bf16) = relu(h @ w1^T + b1)
        gemm_bf16_kernel<<<dim3(DINNER / 128, (QLEN * BSZ) / 128), 256, 0, stream>>>(
            hb, wff1b + (size_t)l * DINNER * DMODEL, nullptr, ffab,
            QLEN * BSZ, DINNER, DMODEL, ff_b1 + (size_t)l * DINNER, 1);

        // ff2(f32) = ff1 @ w2^T + b2
        gemm_bf16_kernel<<<dim3(DMODEL / 128, (QLEN * BSZ) / 128), 256, 0, stream>>>(
            ffab, wff2b + (size_t)l * DMODEL * DINNER, ffo, nullptr,
            QLEN * BSZ, DMODEL, DINNER, ff_b2 + (size_t)l * DMODEL, 0);

        // core = LN(h + ff)   (written to d_out, feeds next layer / final output)
        add_ln_kernel<<<QLEN * BSZ, 256, 0, stream>>>(
            hbuf, ffo, ln2_g + (size_t)l * DMODEL, ln2_b + (size_t)l * DMODEL, core, nullptr);
    }
}
